// EncoderBlock_15745350107390
// MI455X (gfx1250) — matrix operand order repaired and checked by reading, not compiled, not run on hardware
//
#include <hip/hip_runtime.h>
#include <hip/hip_bf16.h>

// ---------------------------------------------------------------------------
// Types for CDNA5 WMMA (gfx1250, wave32)
// ---------------------------------------------------------------------------
typedef __bf16 bf16_t;
typedef __attribute__((ext_vector_type(8)))  __bf16 v8bf;
typedef __attribute__((ext_vector_type(16))) __bf16 v16bf;
typedef __attribute__((ext_vector_type(8)))  float  v8f;

// Pointer types for the async global->LDS builtin (b128 => int vector_size 16)
typedef int v4i_ __attribute__((vector_size(16)));
typedef __attribute__((address_space(1))) v4i_ as1_v4i;
typedef __attribute__((address_space(3))) v4i_ as3_v4i;

// Async global->LDS (CDNA5, ASYNCcnt-tracked) when the toolchain has it.
#if defined(__has_builtin)
#if __has_builtin(__builtin_amdgcn_global_load_async_to_lds_b128)
#define HAVE_ASYNC_LDS 1
#endif
#endif
#ifndef HAVE_ASYNC_LDS
#define HAVE_ASYNC_LDS 0
#endif

static __device__ inline void wait_async_all() {
#if defined(__has_builtin)
#if __has_builtin(__builtin_amdgcn_s_wait_asynccnt)
    __builtin_amdgcn_s_wait_asynccnt(0);
    return;
#else
    asm volatile("s_wait_asynccnt 0" ::: "memory");
    return;
#endif
#else
    asm volatile("s_wait_asynccnt 0" ::: "memory");
#endif
}

// ---------------------------------------------------------------------------
// Problem constants (scoped, no macro collisions)
// ---------------------------------------------------------------------------
constexpr int B_   = 4;
constexpr int T_   = 2048;
constexpr int C_   = 768;
constexpr int H_   = 12;
constexpr int DH_  = 64;
constexpr int BT_  = B_ * T_;   // 8192 rows
constexpr int C4_  = 4 * C_;    // 3072

static __device__ inline v8f wmma_bf16(v16bf a, v16bf b, v8f c) {
    // D = A(16x32 bf16) * B(32x16 bf16) + C(16x16 f32)
    return __builtin_amdgcn_wmma_f32_16x16x32_bf16(
        /*neg_a=*/false, a, /*neg_b=*/false, b,
        /*c_mod=*/(short)0, c, /*reuse_a=*/false, /*reuse_b=*/false);
}

// A fragment (16x32, MxK) from row-major A with leading dim lda.
// lane L (L<16): row m0+L, K = k0+{0..15}
// lane L>=16  : row m0+L-16, K = k0+{16..31}
static __device__ inline v16bf load_afrag(const bf16_t* __restrict__ A, int lda,
                                          int m0, int k0, int lane) {
    int m  = m0 + (lane & 15);
    int kb = k0 + ((lane >> 4) << 4);
    const bf16_t* p = A + (size_t)m * lda + kb;
    v8bf lo = *reinterpret_cast<const v8bf*>(p);
    v8bf hi = *reinterpret_cast<const v8bf*>(p + 8);
    return __builtin_shufflevector(lo, hi, 0,1,2,3,4,5,6,7,8,9,10,11,12,13,14,15);
}

// B fragment (32x16, KxN) from TRANSPOSED storage Bt[N][K] (row-major, ld ldb).
// lane L: column n0+(L&15), K = k0 + (L>=16?16:0) + {0..15}  -> 16 contiguous bf16
static __device__ inline v16bf load_bfrag(const bf16_t* __restrict__ Bt, int ldb,
                                          int n0, int k0, int lane) {
    int n  = n0 + (lane & 15);
    int kb = k0 + ((lane >> 4) << 4);
    return *reinterpret_cast<const v16bf*>(Bt + (size_t)n * ldb + kb);
}

// ---------------------------------------------------------------------------
// Kernel 1: transpose + fp32 -> bf16 convert:  in[b][R][Cc] -> out[b][Cc][R]
// ---------------------------------------------------------------------------
__global__ void transpose_cvt_kernel(const float* __restrict__ in,
                                     bf16_t* __restrict__ out,
                                     int batch, int R, int Cc) {
    size_t idx   = (size_t)blockIdx.x * blockDim.x + threadIdx.x;
    size_t total = (size_t)batch * R * Cc;
    if (idx >= total) return;
    int c = (int)(idx % Cc);
    size_t t = idx / Cc;
    int r = (int)(t % R);
    int b = (int)(t / R);
    out[((size_t)b * Cc + c) * R + r] = (bf16_t)in[idx];
}

// ---------------------------------------------------------------------------
// Kernel 2: LayerNorm (one wave32 per row of 768). Unbiased std, eps on std.
// ---------------------------------------------------------------------------
__global__ __launch_bounds__(256) void ln_kernel(const float* __restrict__ x,
                                                 const float* __restrict__ alpha,
                                                 const float* __restrict__ beta,
                                                 bf16_t* __restrict__ out) {
    int row  = blockIdx.x * 8 + (threadIdx.x >> 5);
    int lane = threadIdx.x & 31;
    const float* xr = x + (size_t)row * C_;
    float vals[24];
    float s = 0.f, ss = 0.f;
#pragma unroll
    for (int i = 0; i < 24; ++i) {
        float v = xr[lane + i * 32];
        vals[i] = v; s += v; ss += v * v;
    }
#pragma unroll
    for (int off = 16; off > 0; off >>= 1) {
        s  += __shfl_xor(s,  off, 32);
        ss += __shfl_xor(ss, off, 32);
    }
    float mean = s * (1.f / (float)C_);
    float var  = (ss - (float)C_ * mean * mean) * (1.f / (float)(C_ - 1)); // ddof=1
    var = var < 0.f ? 0.f : var;
    float inv = 1.f / (sqrtf(var) + 1e-6f); // eps added to std (matches reference)
    bf16_t* orow = out + (size_t)row * C_;
#pragma unroll
    for (int i = 0; i < 24; ++i) {
        int c = lane + i * 32;
        orow[c] = (bf16_t)(alpha[c] * (vals[i] - mean) * inv + beta[c]);
    }
}

// ---------------------------------------------------------------------------
// Kernel 3: templated WMMA GEMM with LDS staging (async global->LDS on CDNA5).
// C[M,N] = A[M,K](bf16) * Bt[N,K](bf16)
// Block: 256 threads = 8 waves (2x4), wave tile 64x64, WG tile 128x256, Kstep 32.
// Per K-step: stage A(128x32, 8KB) + B(256x32, 16KB) in LDS, then ds fragment
// loads feed 16 WMMAs per wave (~87 flop / staged byte).
// MODE 0: QKV scatter (N=2304 -> q,k row-major [T,DH]; v transposed [DH,T])
// MODE 1: out_f32 = acc + bias[n] + resid[m,n]
// MODE 2: out_bf16 = relu(acc + bias[n])
// ---------------------------------------------------------------------------
template <int MODE>
__global__ __launch_bounds__(256) void gemm_bf16_kernel(
    const bf16_t* __restrict__ A, const bf16_t* __restrict__ Bt,
    int N, int K,
    const float* __restrict__ bias, const float* __restrict__ resid,
    float* __restrict__ outF, bf16_t* __restrict__ outB,
    bf16_t* __restrict__ qO, bf16_t* __restrict__ kO, bf16_t* __restrict__ vTO) {
    __shared__ __attribute__((aligned(16))) bf16_t Atile[128 * 32];
    __shared__ __attribute__((aligned(16))) bf16_t Btile[256 * 32];
    const int tid  = threadIdx.x;
    const int lane = tid & 31;
    const int wave = tid >> 5;
    const int mw = (wave & 1) * 64;   // wave-local m origin in WG tile
    const int nw = (wave >> 1) * 64;  // wave-local n origin in WG tile
    const int Mbase = blockIdx.x * 128;
    const int Nbase = blockIdx.y * 256;

    v8f acc[4][4];
#pragma unroll
    for (int i = 0; i < 4; ++i)
#pragma unroll
        for (int j = 0; j < 4; ++j)
            acc[i][j] = (v8f){0.f,0.f,0.f,0.f,0.f,0.f,0.f,0.f};

    for (int k0 = 0; k0 < K; k0 += 32) {
        // ---- stage A tile: 512 chunks of 16B; 2 per thread ----
#pragma unroll
        for (int jj = 0; jj < 2; ++jj) {
            int c   = tid + jj * 256;
            int row = c >> 2, sub = c & 3;
            const bf16_t* gp = A + (size_t)(Mbase + row) * K + k0 + sub * 8;
            bf16_t* lp = &Atile[row * 32 + sub * 8];
#if HAVE_ASYNC_LDS
            __builtin_amdgcn_global_load_async_to_lds_b128(
                (as1_v4i*)gp, (as3_v4i*)lp, 0, 0);
#else
            *reinterpret_cast<v8bf*>(lp) = *reinterpret_cast<const v8bf*>(gp);
#endif
        }
        // ---- stage B tile: 1024 chunks of 16B; 4 per thread ----
#pragma unroll
        for (int jj = 0; jj < 4; ++jj) {
            int c   = tid + jj * 256;
            int row = c >> 2, sub = c & 3;
            const bf16_t* gp = Bt + (size_t)(Nbase + row) * K + k0 + sub * 8;
            bf16_t* lp = &Btile[row * 32 + sub * 8];
#if HAVE_ASYNC_LDS
            __builtin_amdgcn_global_load_async_to_lds_b128(
                (as1_v4i*)gp, (as3_v4i*)lp, 0, 0);
#else
            *reinterpret_cast<v8bf*>(lp) = *reinterpret_cast<const v8bf*>(gp);
#endif
        }
        if (k0 + 32 < K) { // hint next K tile toward L2 (global_prefetch_b8)
            __builtin_prefetch((const void*)(A  + (size_t)(Mbase + (tid >> 1)) * K + k0 + 32), 0, 1);
            __builtin_prefetch((const void*)(Bt + (size_t)(Nbase + tid) * K + k0 + 32), 0, 1);
        }
#if HAVE_ASYNC_LDS
        wait_async_all();
#endif
        __syncthreads();

        // ---- fragments from LDS, 16 WMMAs per wave ----
        v16bf af[4], bfr[4];
#pragma unroll
        for (int i = 0; i < 4; ++i) af[i]  = load_afrag(Atile, 32, mw + i * 16, 0, lane);
#pragma unroll
        for (int j = 0; j < 4; ++j) bfr[j] = load_bfrag(Btile, 32, nw + j * 16, 0, lane);
#pragma unroll
        for (int i = 0; i < 4; ++i)
#pragma unroll
            for (int j = 0; j < 4; ++j)
                acc[i][j] = wmma_bf16(af[i], bfr[j], acc[i][j]);
        __syncthreads();
    }

    // Epilogue. C/D layout: VGPR r, lane L: m_local = r + 8*(L/16), n_local = L%16
    const int half = lane >> 4;
    const int nl   = lane & 15;
#pragma unroll
    for (int i = 0; i < 4; ++i) {
#pragma unroll
        for (int j = 0; j < 4; ++j) {
#pragma unroll
            for (int r = 0; r < 8; ++r) {
                int m = Mbase + mw + i * 16 + r + half * 8;
                int n = Nbase + nw + j * 16 + nl;
                float v = acc[i][j][r];
                if constexpr (MODE == 0) {
                    int b = m >> 11;          // m / T
                    int t = m & (T_ - 1);
                    int which = n / C_;
                    int rem   = n - which * C_;
                    int h = rem >> 6;         // rem / DH
                    int d = rem & (DH_ - 1);
                    bf16_t bv = (bf16_t)v;
                    size_t bh = (size_t)(b * H_ + h);
                    if (which == 0)      qO[(bh * T_ + t) * DH_ + d] = bv;
                    else if (which == 1) kO[(bh * T_ + t) * DH_ + d] = bv;
                    else                 vTO[(bh * DH_ + d) * T_ + t] = bv;
                } else if constexpr (MODE == 1) {
                    outF[(size_t)m * N + n] = v + bias[n] + resid[(size_t)m * N + n];
                } else {
                    float z = v + bias[n];
                    outB[(size_t)m * N + n] = (bf16_t)(z > 0.f ? z : 0.f);
                }
            }
        }
    }
}

// ---------------------------------------------------------------------------
// Kernel 4: flash attention. Block = 128 threads = 4 waves; each wave owns a
// 16-row Q tile; WG covers 64 rows of one (b,h). Grid: (T/64, B*H).
// q,k: [B*H][T][DH] bf16 row-major.  vT: [B*H][DH][T] bf16.
// o: [B*T][C] bf16 (heads concatenated).
// ---------------------------------------------------------------------------
__global__ __launch_bounds__(128) void attn_kernel(
    const bf16_t* __restrict__ q, const bf16_t* __restrict__ k,
    const bf16_t* __restrict__ vT, const int* __restrict__ mask,
    bf16_t* __restrict__ o) {
    __shared__ __attribute__((aligned(16))) bf16_t pbuf[4][16][80]; // per-wave P tile
    const int lane = threadIdx.x & 31;
    const int wave = threadIdx.x >> 5;
    const int bh = blockIdx.y;
    const int b = bh / H_, h = bh % H_;
    const int t0 = blockIdx.x * 64 + wave * 16;
    const int half = lane >> 4, nl = lane & 15;
    const float scale = 0.125f; // 64^-0.5

    const bf16_t* qbase = q  + (size_t)bh * T_ * DH_;
    const bf16_t* kbase = k  + (size_t)bh * T_ * DH_;
    const bf16_t* vbase = vT + (size_t)bh * DH_ * T_;

    v16bf aQ[2];
#pragma unroll
    for (int j = 0; j < 2; ++j) aQ[j] = load_afrag(qbase, DH_, t0, j * 32, lane);

    v8f accO[4];
#pragma unroll
    for (int d = 0; d < 4; ++d) accO[d] = (v8f){0.f,0.f,0.f,0.f,0.f,0.f,0.f,0.f};
    float mst[8], lst[8];
#pragma unroll
    for (int r = 0; r < 8; ++r) { mst[r] = -1e30f; lst[r] = 0.f; }

    for (int s0 = 0; s0 < T_; s0 += 64) {
        // ---- S = Q K^T for a 16x64 tile (4 n-subtiles) ----
        v8f sf[4];
#pragma unroll
        for (int ns = 0; ns < 4; ++ns) {
            v8f a = (v8f){0.f,0.f,0.f,0.f,0.f,0.f,0.f,0.f};
#pragma unroll
            for (int j = 0; j < 2; ++j) {
                v16bf bk = load_bfrag(kbase, DH_, s0 + ns * 16, j * 32, lane);
                a = wmma_bf16(aQ[j], bk, a);
            }
            int scol = s0 + ns * 16 + nl;
#pragma unroll
            for (int r = 0; r < 8; ++r) {
                int t = t0 + r + half * 8;
                float sv = a[r] * scale;
                if (mask[(size_t)t * T_ + scol] == 0) sv = -1e30f;
                a[r] = sv;
            }
            sf[ns] = a;
        }
        // ---- online softmax per row (rows live in 16-lane groups) ----
#pragma unroll
        for (int r = 0; r < 8; ++r) {
            float mx = fmaxf(fmaxf(sf[0][r], sf[1][r]), fmaxf(sf[2][r], sf[3][r]));
#pragma unroll
            for (int off = 8; off > 0; off >>= 1) mx = fmaxf(mx, __shfl_xor(mx, off, 16));
            float mnew = fmaxf(mst[r], mx);
            float corr = __expf(mst[r] - mnew);
            mst[r] = mnew;
            float rs = 0.f;
#pragma unroll
            for (int ns = 0; ns < 4; ++ns) {
                float p = __expf(sf[ns][r] - mnew);
                sf[ns][r] = p; rs += p;
            }
#pragma unroll
            for (int off = 8; off > 0; off >>= 1) rs += __shfl_xor(rs, off, 16);
            lst[r] = lst[r] * corr + rs;
#pragma unroll
            for (int d = 0; d < 4; ++d) accO[d][r] *= corr;
        }
        // ---- P: C-layout -> A-layout via per-wave LDS tile ----
#pragma unroll
        for (int ns = 0; ns < 4; ++ns)
#pragma unroll
            for (int r = 0; r < 8; ++r)
                pbuf[wave][r + half * 8][ns * 16 + nl] = (bf16_t)sf[ns][r];
        v16bf aP[2];
#pragma unroll
        for (int j = 0; j < 2; ++j)
            aP[j] = load_afrag(&pbuf[wave][0][0], 80, 0, j * 32, lane);
        // ---- O += P V  (B from transposed V) ----
#pragma unroll
        for (int d = 0; d < 4; ++d)
#pragma unroll
            for (int j = 0; j < 2; ++j) {
                v16bf bv = load_bfrag(vbase, T_, d * 16, s0 + j * 32, lane);
                accO[d] = wmma_bf16(aP[j], bv, accO[d]);
            }
    }
    // ---- normalize + write heads-concat output ----
#pragma unroll
    for (int d = 0; d < 4; ++d)
#pragma unroll
        for (int r = 0; r < 8; ++r) {
            int t = t0 + r + half * 8;
            float ov = accO[d][r] / lst[r];
            o[((size_t)b * T_ + t) * C_ + h * DH_ + d * 16 + nl] = (bf16_t)ov;
        }
}

// ---------------------------------------------------------------------------
// Host orchestration
// ---------------------------------------------------------------------------
extern "C" void kernel_launch(void* const* d_in, const int* in_sizes, int n_in,
                              void* d_out, int out_size, void* d_ws, size_t ws_size,
                              hipStream_t stream) {
    (void)in_sizes; (void)n_in; (void)out_size; (void)ws_size;
    const float* x      = (const float*)d_in[0];
    const int*   mask   = (const int*)  d_in[1];
    const float* wq     = (const float*)d_in[2];
    const float* wk     = (const float*)d_in[3];
    const float* wv     = (const float*)d_in[4];
    const float* proj_w = (const float*)d_in[5];
    const float* proj_b = (const float*)d_in[6];
    const float* ffn_w1 = (const float*)d_in[7];
    const float* ffn_b1 = (const float*)d_in[8];
    const float* ffn_w2 = (const float*)d_in[9];
    const float* ffn_b2 = (const float*)d_in[10];
    const float* ln1_a  = (const float*)d_in[11];
    const float* ln1_b  = (const float*)d_in[12];
    const float* ln2_a  = (const float*)d_in[13];
    const float* ln2_b  = (const float*)d_in[14];
    float* out = (float*)d_out;

    size_t off = 0;
    auto arena = [&](size_t bytes) -> void* {
        void* p = (char*)d_ws + off;
        off += (bytes + 255) & ~(size_t)255;
        return p;
    };
    bf16_t* wqkvT = (bf16_t*)arena((size_t)3 * C_ * C_ * 2); // [2304][768]
    bf16_t* projT = (bf16_t*)arena((size_t)C_ * C_ * 2);     // [768][768]
    bf16_t* w1T   = (bf16_t*)arena((size_t)C4_ * C_ * 2);    // [3072][768]
    bf16_t* w2T   = (bf16_t*)arena((size_t)C_ * C4_ * 2);    // [768][3072]
    bf16_t* hbuf  = (bf16_t*)arena((size_t)BT_ * C_ * 2);
    bf16_t* qbuf  = (bf16_t*)arena((size_t)BT_ * C_ * 2);
    bf16_t* kbuf  = (bf16_t*)arena((size_t)BT_ * C_ * 2);
    bf16_t* vTbuf = (bf16_t*)arena((size_t)BT_ * C_ * 2);
    bf16_t* obuf  = (bf16_t*)arena((size_t)BT_ * C_ * 2);
    float*  x1    = (float*) arena((size_t)BT_ * C_ * 4);
    bf16_t* h2    = (bf16_t*)arena((size_t)BT_ * C_ * 2);
    bf16_t* f1    = (bf16_t*)arena((size_t)BT_ * C4_ * 2);

    auto tgrid = [](size_t total) { return dim3((unsigned)((total + 255) / 256)); };

    // --- weight prep (bf16, transposed [N][K]) ---
    transpose_cvt_kernel<<<tgrid((size_t)H_*C_*DH_), 256, 0, stream>>>(wq, wqkvT,               H_, C_, DH_);
    transpose_cvt_kernel<<<tgrid((size_t)H_*C_*DH_), 256, 0, stream>>>(wk, wqkvT + (size_t)C_*C_,   H_, C_, DH_);
    transpose_cvt_kernel<<<tgrid((size_t)H_*C_*DH_), 256, 0, stream>>>(wv, wqkvT + (size_t)2*C_*C_, H_, C_, DH_);
    transpose_cvt_kernel<<<tgrid((size_t)C_*C_),     256, 0, stream>>>(proj_w, projT, 1, C_, C_);
    transpose_cvt_kernel<<<tgrid((size_t)C_*C4_),    256, 0, stream>>>(ffn_w1, w1T,   1, C_, C4_);
    transpose_cvt_kernel<<<tgrid((size_t)C4_*C_),    256, 0, stream>>>(ffn_w2, w2T,   1, C4_, C_);

    // --- residual 1: LN1 -> fused QKV -> flash attention -> proj + x ---
    ln_kernel<<<dim3(BT_ / 8), 256, 0, stream>>>(x, ln1_a, ln1_b, hbuf);

    gemm_bf16_kernel<0><<<dim3(BT_ / 128, (3 * C_) / 256), 256, 0, stream>>>(
        hbuf, wqkvT, 3 * C_, C_, nullptr, nullptr, nullptr, nullptr,
        qbuf, kbuf, vTbuf);

    attn_kernel<<<dim3(T_ / 64, B_ * H_), 128, 0, stream>>>(qbuf, kbuf, vTbuf, mask, obuf);

    gemm_bf16_kernel<1><<<dim3(BT_ / 128, C_ / 256), 256, 0, stream>>>(
        obuf, projT, C_, C_, proj_b, x, x1, nullptr, nullptr, nullptr, nullptr);

    // --- residual 2: LN2 -> FFN1(+ReLU) -> FFN2 + x1 ---
    ln_kernel<<<dim3(BT_ / 8), 256, 0, stream>>>(x1, ln2_a, ln2_b, h2);

    gemm_bf16_kernel<2><<<dim3(BT_ / 128, C4_ / 256), 256, 0, stream>>>(
        h2, w1T, C4_, C_, ffn_b1, nullptr, nullptr, f1, nullptr, nullptr, nullptr);

    gemm_bf16_kernel<1><<<dim3(BT_ / 128, C_ / 256), 256, 0, stream>>>(
        f1, w2T, C_, C4_, ffn_b2, x1, out, nullptr, nullptr, nullptr, nullptr);
}
